// GMN_9268539425140
// MI455X (gfx1250) — matrix operand order repaired and checked by reading, not compiled, not run on hardware
//
#include <hip/hip_runtime.h>
#include <hip/hip_bf16.h>

// ---------------- problem constants (from reference) ----------------
static constexpr int   Bsz   = 256;
static constexpr int   NPG   = 256;
static constexpr int   NTOT  = Bsz * NPG;       // 65536
static constexpr int   Eg    = 1048576;
static constexpr int   FIN   = 64;
static constexpr int   HID   = 128;
static constexpr int   HEADS = 5;
static constexpr int   K1    = 32;
static constexpr int   K2    = 8;
static constexpr int   MEMH  = 100;
static constexpr int   NCLS  = 12;
static constexpr float EPSF  = 1e-8f;

typedef __attribute__((ext_vector_type(16))) __bf16 v16bf;
typedef __attribute__((ext_vector_type(8)))  float  v8f;

// ============ WMMA helpers (bf16 in, f32 accum), 16x16x32 tiles ============
// A fragment k order used here (16-bit A 16x32, not AMD's table order): lane idx = M, element e -> K = half*16+e
// B fragment k order used here (16-bit B 32x16, same as A): lane idx = N, element e -> K = half*16+e
// C/D f32: element r -> M = r + 8*half, lane idx = N

// NT output tiles sharing one A fragment; B staged in LDS as bf16, layout [n][k],
// padded row stride ldp (conflict-free, 16B aligned -> ds_load_b128).
template <int NT, bool TRANSA>
__device__ __forceinline__ void wmma_ldsB_tiles(const float* __restrict__ A, int lda,
                                                const __bf16* __restrict__ Bl, int ldp,
                                                int row0, int col0, int K, v8f* acc) {
  const int lane = threadIdx.x & 31;
  const int idx  = lane & 15;
  const int half = lane >> 4;
  for (int k0 = 0; k0 < K; k0 += 32) {
    v16bf a;
#pragma unroll
    for (int e = 0; e < 16; ++e) {
      const int ka = k0 + half * 16 + e;
      const float av = TRANSA ? A[(size_t)ka * lda + (row0 + idx)]
                              : A[(size_t)(row0 + idx) * lda + ka];
      a[e] = (__bf16)av;
    }
    const int kb = k0 + half * 16;
#pragma unroll
    for (int t = 0; t < NT; ++t) {
      const int n = col0 + t * 16 + idx;
      union { v16bf v; float4 f[2]; } ub;
      const float4* p = (const float4*)&Bl[n * ldp + kb];   // 32B, 16B-aligned
      ub.f[0] = p[0]; ub.f[1] = p[1];
      acc[t] = __builtin_amdgcn_wmma_f32_16x16x32_bf16(false, a, false, ub.v,
                                                       (short)0, acc[t], false, false);
    }
  }
}

// NT output tiles, B from global f32. TRANSB=0: B[k*ldb+n], TRANSB=1: B[n*ldb+k]
template <int NT, bool TRANSA, bool TRANSB>
__device__ __forceinline__ void wmma_tiles_g(const float* __restrict__ A, int lda,
                                             const float* __restrict__ B, int ldb,
                                             int row0, int col0, int K, v8f* acc) {
  const int lane = threadIdx.x & 31;
  const int idx  = lane & 15;
  const int half = lane >> 4;
  for (int k0 = 0; k0 < K; k0 += 32) {
    v16bf a;
#pragma unroll
    for (int e = 0; e < 16; ++e) {
      const int ka = k0 + half * 16 + e;
      const float av = TRANSA ? A[(size_t)ka * lda + (row0 + idx)]
                              : A[(size_t)(row0 + idx) * lda + ka];
      a[e] = (__bf16)av;
    }
#pragma unroll
    for (int t = 0; t < NT; ++t) {
      const int n = col0 + t * 16 + idx;
      v16bf b;
#pragma unroll
      for (int e = 0; e < 16; ++e) {
        const int kb = k0 + half * 16 + e;
        const float bv = TRANSB ? B[(size_t)n * ldb + kb] : B[(size_t)kb * ldb + n];
        b[e] = (__bf16)bv;
      }
      acc[t] = __builtin_amdgcn_wmma_f32_16x16x32_bf16(false, a, false, b,
                                                       (short)0, acc[t], false, false);
    }
  }
}

// Stage a KxN row-major f32 weight matrix into LDS as bf16, layout [n][k], stride ldp.
// Global reads are fully coalesced (linear over W).
template <int K, int N>
__device__ __forceinline__ void stage_weight_bf16(const float* __restrict__ W,
                                                  __bf16* __restrict__ Bl) {
  constexpr int LDP = K + 8;
  for (int i = threadIdx.x; i < N * K; i += blockDim.x) {
    const int k = i / N, n = i - k * N;
    Bl[n * LDP + k] = (__bf16)W[i];
  }
}

// Stage a rowsxK row-major f32 matrix (rows = logical N, e.g. keys) into LDS bf16 [n][k].
__device__ __forceinline__ void stage_rows_bf16(const float* __restrict__ src,
                                                __bf16* __restrict__ Bl,
                                                int rows, int K, int ldp) {
  for (int i = threadIdx.x; i < rows * K; i += blockDim.x) {
    const int n = i / K, k = i - n * K;
    Bl[n * ldp + k] = (__bf16)src[i];
  }
}

// ---------------- edge scatter: aggr[dst] += x[src] ----------------
template <int F>
__global__ void scatter_add_kernel(const float* __restrict__ xin,
                                   const int* __restrict__ src,
                                   const int* __restrict__ dst,
                                   float* __restrict__ aggr) {
  const int edge = blockIdx.x * (blockDim.x >> 5) + (threadIdx.x >> 5);
  const int lane = threadIdx.x & 31;
  if (edge >= Eg) return;
  if (edge + 8 < Eg) {                       // prefetch next edge's gather row
    const int sn = src[edge + 8];
    __builtin_prefetch(&xin[(size_t)sn * F], 0, 1);   // -> global_prefetch_b8
  }
  const int s = src[edge];
  const int d = dst[edge];
#pragma unroll
  for (int f = lane; f < F; f += 32)
    atomicAdd(&aggr[(size_t)d * F + f], xin[(size_t)s * F + f]);
}

// ------- graph conv GEMM: out = relu(aggr@Wrel + b + x@Wroot), weights in LDS -------
// Block: 256 thr = 8 waves; wave w -> row strip tm = 4*blk + (w>>1), cols (w&1)*64..+63
template <int K>
__global__ void graphconv_gemm_kernel(const float* __restrict__ aggr,
                                      const float* __restrict__ xin,
                                      const float* __restrict__ Wrel,
                                      const float* __restrict__ brel,
                                      const float* __restrict__ Wroot,
                                      float* __restrict__ out) {
  constexpr int LDP = K + 8;
  __shared__ __bf16 Bl[HID * LDP];
  const int w = threadIdx.x >> 5;
  const int row0 = (blockIdx.x * 4 + (w >> 1)) * 16;
  const int col0 = (w & 1) * 64;
  v8f acc[4] = {};
  stage_weight_bf16<K, HID>(Wrel, Bl);
  __syncthreads();
  wmma_ldsB_tiles<4, false>(aggr, K, Bl, LDP, row0, col0, K, acc);
  __syncthreads();
  stage_weight_bf16<K, HID>(Wroot, Bl);
  __syncthreads();
  wmma_ldsB_tiles<4, false>(xin, K, Bl, LDP, row0, col0, K, acc);

  const int lane = threadIdx.x & 31;
  const int nn = lane & 15, half = lane >> 4;
#pragma unroll
  for (int t = 0; t < 4; ++t) {
    const int col = col0 + t * 16 + nn;
    const float bias = brel[col];
#pragma unroll
    for (int r = 0; r < 8; ++r) {
      const int m = row0 + r + 8 * half;
      const float v = acc[t][r] + bias;
      out[(size_t)m * HID + col] = v > 0.f ? v : 0.f;
    }
  }
}

// ---------------- batch norm ----------------
__global__ void bn_stats_kernel(const float* __restrict__ q, float* __restrict__ stat) {
  const int col = threadIdx.x;                 // 128
  const int rows = NTOT / gridDim.x;
  const int r0 = blockIdx.x * rows;
  float s = 0.f, s2 = 0.f;
  for (int r = 0; r < rows; ++r) {
    const float v = q[(size_t)(r0 + r) * HID + col];
    s += v; s2 += v * v;
  }
  atomicAdd(&stat[col], s);
  atomicAdd(&stat[HID + col], s2);
}

__global__ void bn_apply_kernel(float* __restrict__ q, const float* __restrict__ stat,
                                const float* __restrict__ gamma, const float* __restrict__ beta) {
  const int idx = blockIdx.x * blockDim.x + threadIdx.x;   // NTOT*HID exact
  const int col = idx & (HID - 1);
  const float mu  = stat[col] * (1.f / NTOT);
  const float var = stat[HID + col] * (1.f / NTOT) - mu * mu;
  const float r   = rsqrtf(var + 1e-5f);
  q[idx] = (q[idx] - mu) * r * gamma[col] + beta[col];
}

// ---------------- row squared-norms ----------------
__global__ void rowsq_kernel(const float* __restrict__ A, float* __restrict__ out,
                             int rows, int K) {
  const int g = blockIdx.x * blockDim.x + threadIdx.x;
  if (g >= rows) return;
  const float* r = A + (size_t)g * K;
  float s = 0.f;
  for (int k = 0; k < K; ++k) s += r[k] * r[k];
  out[g] = s;
}

// -------- qk GEMM, keys panel in LDS, fused Cauchy epilogue --------
// craw[g, hk] = 1/(1 + max(qsq[g]+ksq[hk]-2*(Q@keys^T)[g,hk], 1e-12))
// Block: 8 waves; wave w -> strip tm = 4*blk + (w>>1), cols (w&1)*80..+79 (5 tiles)
__global__ void qkgemm_kernel(const float* __restrict__ Q, const float* __restrict__ keys,
                              const float* __restrict__ qsq, const float* __restrict__ ksq,
                              float* __restrict__ craw) {
  constexpr int LDP = HID + 8;                  // 136
  __shared__ __bf16 Bl[HEADS * K1 * LDP];       // 160*136 bf16 = 43.5 KB
  const int w = threadIdx.x >> 5;
  const int tm = blockIdx.x * 4 + (w >> 1);
  const int col0 = (w & 1) * 80;
  v8f acc[5] = {};
  stage_rows_bf16(keys, Bl, HEADS * K1, HID, LDP);
  __syncthreads();
  wmma_ldsB_tiles<5, false>(Q, HID, Bl, LDP, tm * 16, col0, HID, acc);

  const int lane = threadIdx.x & 31;
  const int nn = lane & 15, half = lane >> 4;
#pragma unroll
  for (int t = 0; t < 5; ++t) {
    const int n = col0 + t * 16 + nn;
    const float ks = ksq[n];
#pragma unroll
    for (int r = 0; r < 8; ++r) {
      const int m = tm * 16 + r + 8 * half;
      const float d2 = qsq[m] + ks - 2.f * acc[t][r];
      craw[(size_t)m * (HEADS * K1) + n] =
          __builtin_amdgcn_rcpf(1.f + fmaxf(d2, 1e-12f));
    }
  }
}

// ---------------- per-head normalize, head-combine, softmax over k ----------------
template <int NH, int KK>
__global__ void head_softmax_kernel(const float* __restrict__ craw,
                                    const float* __restrict__ convw,
                                    float* __restrict__ Cout, int nrows) {
  const int g = blockIdx.x * blockDim.x + threadIdx.x;
  if (g >= nrows) return;
  float comb[KK];
#pragma unroll
  for (int k = 0; k < KK; ++k) comb[k] = 0.f;
  for (int h = 0; h < NH; ++h) {
    const float* row = craw + (size_t)g * (NH * KK) + h * KK;
    float s = 0.f;
    for (int k = 0; k < KK; ++k) s += row[k];
    const float w = convw[h] / s;
    for (int k = 0; k < KK; ++k) comb[k] += w * row[k];
  }
  float mx = comb[0];
#pragma unroll
  for (int k = 1; k < KK; ++k) mx = fmaxf(mx, comb[k]);
  float se = 0.f;
#pragma unroll
  for (int k = 0; k < KK; ++k) { comb[k] = expf(comb[k] - mx); se += comb[k]; }
  const float inv = 1.f / se;
#pragma unroll
  for (int k = 0; k < KK; ++k) Cout[(size_t)g * KK + k] = comb[k] * inv;
}

// ---------------- column sums over n (per batch) ----------------
__global__ void colsum_kernel(const float* __restrict__ C, float* __restrict__ cn,
                              int npg, int KK) {
  const int b = blockIdx.x, k = threadIdx.x;
  float s = 0.f;
  for (int n = 0; n < npg; ++n) s += C[((size_t)b * npg + n) * KK + k];
  cn[b * KK + k] = s;
}

// ---------------- KL regularizer (fused per row, atomic accumulate) ----------------
template <int KK>
__global__ void kl_kernel(const float* __restrict__ C, const float* __restrict__ cn,
                          int npg, int nrows, float* __restrict__ klacc) {
  const int g = blockIdx.x * blockDim.x + threadIdx.x;
  if (g >= nrows) return;
  const int b = g / npg;
  float P[KK];
  float pn = 0.f;
#pragma unroll
  for (int k = 0; k < KK; ++k) {
    const float c = C[(size_t)g * KK + k];
    const float p = c * c / (cn[b * KK + k] + EPSF);
    P[k] = p; pn += p;
  }
  pn += EPSF;
  float s = 0.f;
#pragma unroll
  for (int k = 0; k < KK; ++k) {
    const float p = P[k] / pn;
    const float c = C[(size_t)g * KK + k];
    s += p * (logf(p + EPSF) - logf(c + EPSF));
  }
  atomicAdd(klacc, 100.f * s);
}

// -------- V = C^T @ Q per batch (32x128 = 32x256 @ 256x128), NT=4 blocking --------
// wave id -> b = wid>>2 ; tm = (wid>>1)&1 ; col group = wid&1 (64 cols)
__global__ void vgemm_kernel(const float* __restrict__ Cmat, const float* __restrict__ Q,
                             float* __restrict__ V) {
  const int wid = blockIdx.x * (blockDim.x >> 5) + (threadIdx.x >> 5);
  const int b  = wid >> 2;
  const int tm = (wid >> 1) & 1;
  const int col0 = (wid & 1) * 64;
  v8f acc[4] = {};
  wmma_tiles_g<4, true, false>(Cmat + (size_t)b * NPG * K1, K1,
                               Q + (size_t)b * NPG * HID, HID,
                               tm * 16, col0, NPG, acc);
  const int lane = threadIdx.x & 31;
  const int nn = lane & 15, half = lane >> 4;
#pragma unroll
  for (int t = 0; t < 4; ++t) {
    const int col = col0 + t * 16 + nn;
#pragma unroll
    for (int r = 0; r < 8; ++r) {
      const int m = tm * 16 + r + 8 * half;              // 0..31
      V[((size_t)b * K1 + m) * HID + col] = acc[t][r];
    }
  }
}

// ---------------- plain linear + leaky relu ----------------
__global__ void lin_leaky_kernel(const float* __restrict__ A, const float* __restrict__ W,
                                 const float* __restrict__ bias, float* __restrict__ out,
                                 int M, int K, int N) {
  const int idx = blockIdx.x * blockDim.x + threadIdx.x;
  if (idx >= M * N) return;
  const int m = idx / N, n = idx % N;
  float acc = bias[n];
  for (int k = 0; k < K; ++k) acc += A[(size_t)m * K + k] * W[(size_t)k * N + n];
  out[idx] = acc > 0.f ? acc : 0.01f * acc;
}

// ---------------- qk for mem-conv-1 (tiny, ragged K=100) ----------------
__global__ void qk2_kernel(const float* __restrict__ Qm, const float* __restrict__ keys,
                           const float* __restrict__ qsq2, const float* __restrict__ ksq2,
                           float* __restrict__ craw2) {
  const int idx = blockIdx.x * blockDim.x + threadIdx.x;   // (B*K1) * (HEADS*K2)
  if (idx >= Bsz * K1 * HEADS * K2) return;
  const int g = idx / (HEADS * K2), c = idx % (HEADS * K2);
  const float* qr = Qm + (size_t)g * MEMH;
  const float* kr = keys + (size_t)c * MEMH;
  float acc = 0.f;
  for (int f = 0; f < MEMH; ++f) acc += qr[f] * kr[f];
  const float d2 = qsq2[g] + ksq2[c] - 2.f * acc;
  craw2[idx] = __builtin_amdgcn_rcpf(1.f + fmaxf(d2, 1e-12f));
}

// ---------------- V2[b,k2,f] = sum_n C2[b,n,k2] * Q1m[b,n,f] ----------------
__global__ void v2_kernel(const float* __restrict__ C2, const float* __restrict__ Q1m,
                          float* __restrict__ V2) {
  const int idx = blockIdx.x * blockDim.x + threadIdx.x;
  if (idx >= Bsz * K2 * MEMH) return;
  const int b = idx / (K2 * MEMH);
  const int rem = idx % (K2 * MEMH);
  const int k2 = rem / MEMH, f = rem % MEMH;
  float s = 0.f;
  for (int n = 0; n < K1; ++n)
    s += C2[((size_t)b * K1 + n) * K2 + k2] * Q1m[((size_t)b * K1 + n) * MEMH + f];
  V2[idx] = s;
}

// ---------------- h = mean over K2 ----------------
__global__ void hmean_kernel(const float* __restrict__ q2, float* __restrict__ h) {
  const int idx = blockIdx.x * blockDim.x + threadIdx.x;
  if (idx >= Bsz * MEMH) return;
  const int b = idx / MEMH, f = idx % MEMH;
  float s = 0.f;
  for (int k = 0; k < K2; ++k) s += q2[((size_t)b * K2 + k) * MEMH + f];
  h[idx] = s * (1.f / K2);
}

// ---------------- final MLP: 100 -> 50 (leaky) -> 12 ----------------
__global__ void mlp_kernel(const float* __restrict__ h, const float* __restrict__ W1,
                           const float* __restrict__ b1, const float* __restrict__ W2,
                           const float* __restrict__ b2, float* __restrict__ yh) {
  __shared__ float hid[50];
  const int b = blockIdx.x;
  const int t = threadIdx.x;                   // 64 threads
  if (t < 50) {
    float a = b1[t];
    for (int k = 0; k < MEMH; ++k) a += h[(size_t)b * MEMH + k] * W1[k * 50 + t];
    hid[t] = a > 0.f ? a : 0.01f * a;
  }
  __syncthreads();
  if (t < NCLS) {
    float a = b2[t];
    for (int k = 0; k < 50; ++k) a += hid[k] * W2[k * NCLS + t];
    yh[b * NCLS + t] = a;
  }
}

__global__ void finalize_kernel(const float* __restrict__ klacc, float* __restrict__ out) {
  out[Bsz * NCLS] = klacc[0] / (float)NTOT;
}

// ======================================================================
extern "C" void kernel_launch(void* const* d_in, const int* in_sizes, int n_in,
                              void* d_out, int out_size, void* d_ws, size_t ws_size,
                              hipStream_t stream) {
  const float* x       = (const float*)d_in[0];
  const int*   ei      = (const int*)  d_in[1];
  const int*   src     = ei;
  const int*   dst     = ei + Eg;
  // d_in[2] (batch) unused by the reference math (reshape implies grouping)
  const float* W_rel0  = (const float*)d_in[3];
  const float* b_rel0  = (const float*)d_in[4];
  const float* W_root0 = (const float*)d_in[5];
  const float* W_rel1  = (const float*)d_in[6];
  const float* b_rel1  = (const float*)d_in[7];
  const float* W_root1 = (const float*)d_in[8];
  const float* bn_g    = (const float*)d_in[9];
  const float* bn_b    = (const float*)d_in[10];
  const float* keys0   = (const float*)d_in[11];   // (5,32,128) flat = (160,128)
  const float* convw0  = (const float*)d_in[12];
  const float* lin_W0  = (const float*)d_in[13];
  const float* lin_b0  = (const float*)d_in[14];
  const float* keys1   = (const float*)d_in[15];   // (5,8,100) flat = (40,100)
  const float* convw1  = (const float*)d_in[16];
  const float* lin_W1  = (const float*)d_in[17];
  const float* lin_b1  = (const float*)d_in[18];
  const float* mlp_W1  = (const float*)d_in[19];
  const float* mlp_b1  = (const float*)d_in[20];
  const float* mlp_W2  = (const float*)d_in[21];
  const float* mlp_b2  = (const float*)d_in[22];
  float* out = (float*)d_out;

  // -------- workspace layout (floats) --------
  float* ws = (float*)d_ws;
  size_t o = 0;
  float* aggr  = ws + o; o += (size_t)NTOT * HID;
  float* q0    = ws + o; o += (size_t)NTOT * HID;
  float* q1    = ws + o; o += (size_t)NTOT * HID;
  float* craw  = ws + o; o += (size_t)NTOT * (HEADS * K1);
  float* Cmat  = ws + o; o += (size_t)NTOT * K1;
  float* Vbuf  = ws + o; o += (size_t)Bsz * K1 * HID;
  float* out1  = ws + o; o += (size_t)Bsz * K1 * MEMH;
  float* qsq   = ws + o; o += NTOT;
  float* ksq   = ws + o; o += HEADS * K1;
  float* craw2 = ws + o; o += (size_t)Bsz * K1 * HEADS * K2;
  float* c2    = ws + o; o += (size_t)Bsz * K1 * K2;
  float* qsq2  = ws + o; o += Bsz * K1;
  float* ksq2  = ws + o; o += HEADS * K2;
  float* V2    = ws + o; o += (size_t)Bsz * K2 * MEMH;
  float* q2b   = ws + o; o += (size_t)Bsz * K2 * MEMH;
  float* hbuf  = ws + o; o += Bsz * MEMH;
  float* stat  = ws + o; o += 2 * HID;          // small accumulators, contiguous
  float* cn    = ws + o; o += Bsz * K1;
  float* cn2   = ws + o; o += Bsz * K2;
  float* klacc = ws + o; o += 2;

  // -------- zero accumulators --------
  hipMemsetAsync(aggr, 0, (size_t)NTOT * HID * sizeof(float), stream);
  hipMemsetAsync(stat, 0, (2 * HID + Bsz * K1 + Bsz * K2 + 2) * sizeof(float), stream);

  // -------- graph conv 0 --------
  scatter_add_kernel<FIN><<<Eg / 8, 256, 0, stream>>>(x, src, dst, aggr);
  graphconv_gemm_kernel<FIN><<<(NTOT / 16) / 4, 256, 0, stream>>>(
      aggr, x, W_rel0, b_rel0, W_root0, q0);

  // -------- graph conv 1 --------
  hipMemsetAsync(aggr, 0, (size_t)NTOT * HID * sizeof(float), stream);
  scatter_add_kernel<HID><<<Eg / 8, 256, 0, stream>>>(q0, src, dst, aggr);
  graphconv_gemm_kernel<HID><<<(NTOT / 16) / 4, 256, 0, stream>>>(
      aggr, q0, W_rel1, b_rel1, W_root1, q1);

  // -------- batch norm (in place on q1) --------
  bn_stats_kernel<<<256, HID, 0, stream>>>(q1, stat);
  bn_apply_kernel<<<(NTOT * HID) / 256, 256, 0, stream>>>(q1, stat, bn_g, bn_b);

  // -------- mem conv 0 --------
  rowsq_kernel<<<NTOT / 256, 256, 0, stream>>>(q1, qsq, NTOT, HID);
  rowsq_kernel<<<1, 256, 0, stream>>>(keys0, ksq, HEADS * K1, HID);
  qkgemm_kernel<<<(NTOT / 16) / 4, 256, 0, stream>>>(q1, keys0, qsq, ksq, craw);
  head_softmax_kernel<HEADS, K1><<<NTOT / 256, 256, 0, stream>>>(craw, convw0, Cmat, NTOT);
  colsum_kernel<<<Bsz, K1, 0, stream>>>(Cmat, cn, NPG, K1);
  kl_kernel<K1><<<NTOT / 256, 256, 0, stream>>>(Cmat, cn, NPG, NTOT, klacc);
  vgemm_kernel<<<Bsz * 4 / 8, 256, 0, stream>>>(Cmat, q1, Vbuf);
  lin_leaky_kernel<<<(Bsz * K1 * MEMH + 255) / 256, 256, 0, stream>>>(
      Vbuf, lin_W0, lin_b0, out1, Bsz * K1, HID, MEMH);

  // -------- mem conv 1 (tiny) --------
  rowsq_kernel<<<(Bsz * K1 + 255) / 256, 256, 0, stream>>>(out1, qsq2, Bsz * K1, MEMH);
  rowsq_kernel<<<1, 64, 0, stream>>>(keys1, ksq2, HEADS * K2, MEMH);
  qk2_kernel<<<(Bsz * K1 * HEADS * K2 + 255) / 256, 256, 0, stream>>>(
      out1, keys1, qsq2, ksq2, craw2);
  head_softmax_kernel<HEADS, K2><<<(Bsz * K1 + 255) / 256, 256, 0, stream>>>(
      craw2, convw1, c2, Bsz * K1);
  colsum_kernel<<<Bsz, K2, 0, stream>>>(c2, cn2, K1, K2);
  kl_kernel<K2><<<(Bsz * K1 + 255) / 256, 256, 0, stream>>>(c2, cn2, K1, Bsz * K1, klacc);
  v2_kernel<<<(Bsz * K2 * MEMH + 255) / 256, 256, 0, stream>>>(c2, out1, V2);
  lin_leaky_kernel<<<(Bsz * K2 * MEMH + 255) / 256, 256, 0, stream>>>(
      V2, lin_W1, lin_b1, q2b, Bsz * K2, MEMH, MEMH);

  // -------- head --------
  hmean_kernel<<<(Bsz * MEMH + 255) / 256, 256, 0, stream>>>(q2b, hbuf);
  mlp_kernel<<<Bsz, 64, 0, stream>>>(hbuf, mlp_W1, mlp_b1, mlp_W2, mlp_b2, out);
  finalize_kernel<<<1, 1, 0, stream>>>(klacc, out);
}
